// MultiHeadAttention_16346645529223
// MI455X (gfx1250) — matrix operand order repaired and checked by reading, not compiled, not run on hardware
//
#include <hip/hip_runtime.h>
#include <hip/hip_bf16.h>

// Problem constants (match reference)
#define BB   4
#define SQL  2048
#define SKVL 2048
#define DDIM 1024
#define NH   16
#define DH   64

typedef __bf16 bf16_t;
typedef __attribute__((ext_vector_type(16))) __bf16 v16bf;
typedef __attribute__((ext_vector_type(8)))  __bf16 v8bf;
typedef __attribute__((ext_vector_type(8)))  float  v8f;
typedef __attribute__((ext_vector_type(4)))  float  v4f;
typedef __attribute__((ext_vector_type(4)))  unsigned int v4u;
typedef __attribute__((ext_vector_type(8)))  int    v8i;
typedef __attribute__((ext_vector_type(4)))  int    v4i;

#if __has_builtin(__builtin_amdgcn_tensor_load_to_lds) && \
    __has_builtin(__builtin_amdgcn_s_wait_tensorcnt)
#define USE_TDM 1
#else
#define USE_TDM 0
#endif

static __device__ __forceinline__ v8f wmma_bf16(v16bf a, v16bf b, v8f c) {
  // D = A(16x32) * B(32x16) + C, f32 accumulate
  return __builtin_amdgcn_wmma_f32_16x16x32_bf16(false, a, false, b, (short)0, c,
                                                 false, false);
}

// A-operand fragment (16x32 bf16), source row-major, rows = m, k contiguous.
// CDNA5 layout: lane = 16g + m; VGPR0..3 hold k=8g..8g+7, VGPR4..7 hold k=16+8g..23+8g.
static __device__ __forceinline__ v16bf load_a_frag(const bf16_t* base, int ld) {
  int l = threadIdx.x & 31;
  const bf16_t* p = base + (size_t)(l & 15) * ld + ((l >> 4) << 3);
  v8bf lo = *(const v8bf*)p;          // k = 8g .. 8g+7
  v8bf hi = *(const v8bf*)(p + 16);   // k = 16+8g .. 23+8g
  return __builtin_shufflevector(lo, hi, 0,1,2,3,4,5,6,7,8,9,10,11,12,13,14,15);
}

// B-operand fragment (32x16 bf16), source row-major as B^T (rows = n, k contiguous).
// CDNA5 layout: lane = 16g + n holds k = 8g..8g+7, then k = 16+8g..23+8g.
static __device__ __forceinline__ v16bf load_b_frag(const bf16_t* base, int ld) {
  int l = threadIdx.x & 31;
  const bf16_t* p = base + (size_t)(l & 15) * ld + ((l >> 4) << 3);
  v8bf lo = *(const v8bf*)p;
  v8bf hi = *(const v8bf*)(p + 16);
  return __builtin_shufflevector(lo, hi, 0,1,2,3,4,5,6,7,8,9,10,11,12,13,14,15);
}

#if USE_TDM
// LDS byte offset of a shared-memory pointer (generic LDS addr low 32 bits).
static __device__ __forceinline__ unsigned lds_off(const void* p) {
  return (unsigned)(uintptr_t)p;
}

// Issue a 2-D TENSOR_LOAD_TO_LDS per CDNA5 D# spec (data_size = 2 bytes).
// tile rows land contiguously in LDS (tile_d0 elems per row).
static __device__ __forceinline__ void tdm_2d_load(
    unsigned lds, const void* gaddr, unsigned tensor_d0, unsigned tensor_d1,
    unsigned stride0, unsigned tile_d0, unsigned tile_d1) {
  unsigned long long ga = (unsigned long long)(uintptr_t)gaddr;
  v4u g0;
  g0[0] = 1u;                                   // count=1, user descriptor
  g0[1] = lds;                                  // lds_addr
  g0[2] = (unsigned)(ga & 0xFFFFFFFFu);         // global_addr[31:0]
  g0[3] = (unsigned)((ga >> 32) & 0x01FFFFFFu)  // global_addr[56:32]
          | 0x80000000u;                        // type=2 ("image")
  v8i g1;
  g1[0] = (int)(1u << 16);                      // data_size=1 -> 2 bytes
  g1[1] = (int)((tensor_d0 & 0xFFFFu) << 16);   // tensor_dim0[15:0]
  g1[2] = (int)((tensor_d1 & 0xFFFFu) << 16);   // tensor_dim1[15:0]
  g1[3] = (int)((tile_d0 & 0xFFFFu) << 16);     // tile_dim0
  g1[4] = (int)(tile_d1 & 0xFFFFu);             // tile_dim1 (tile_dim2=0)
  g1[5] = (int)stride0;                         // tensor_dim0_stride[31:0]
  g1[6] = 0;
  g1[7] = 0;
  v4i z4 = {0, 0, 0, 0};
#if defined(__clang_major__) && __clang_major__ >= 23
  v8i z8 = {0, 0, 0, 0, 0, 0, 0, 0};
  __builtin_amdgcn_tensor_load_to_lds(g0, g1, z4, z4, z8, 0);
#else
  __builtin_amdgcn_tensor_load_to_lds(g0, g1, z4, z4, 0);
#endif
}
#endif  // USE_TDM

// ---------------- elementwise f32 -> bf16 ----------------
__global__ void cvt_f32_bf16(const float* __restrict__ in, bf16_t* __restrict__ out,
                             int n) {
  int i = blockIdx.x * blockDim.x + threadIdx.x;
  if (i < n) out[i] = (bf16_t)in[i];
}

// ---------------- weight transpose + convert: out[h][e][d] = in[h][d][e] -----
__global__ void transpose_w(const float* __restrict__ in, bf16_t* __restrict__ out,
                            int Hn, int Din, int En) {
  int i = blockIdx.x * blockDim.x + threadIdx.x;
  int total = Hn * Din * En;
  if (i >= total) return;
  int d = i % Din;
  int t = i / Din;
  int e = t % En;
  int h = t / En;
  out[i] = (bf16_t)in[((size_t)h * Din + d) * En + e];
}

// ---------------- fused QKV projection (WMMA GEMM) ----------------
// Q,K: transposed orientation (A = Wt rows, B = X rows) -> packed row-major store.
// V  : normal orientation    (A = X rows, B = Wt rows) -> packed V^T store.
__global__ __launch_bounds__(128) void qkv_proj(
    const bf16_t* __restrict__ Xq, const bf16_t* __restrict__ Xv,
    const bf16_t* __restrict__ Wtq, const bf16_t* __restrict__ Wtk,
    const bf16_t* __restrict__ Wtv,
    const float* __restrict__ bq, const float* __restrict__ bk,
    const float* __restrict__ bv,
    bf16_t* __restrict__ Qo, bf16_t* __restrict__ Ko, bf16_t* __restrict__ Vto) {
  int wave = threadIdx.x >> 5;
  int l = threadIdx.x & 31, n = l & 15, g = l >> 4;
  int which = blockIdx.z % 3;  // 0=Q,1=K,2=V
  int b = blockIdx.z / 3;
  int h = blockIdx.y;
  int s0 = blockIdx.x * 64 + wave * 16;

  const bf16_t* X = (which == 0 ? Xq : Xv) + (size_t)b * SQL * DDIM;
  const bf16_t* Wt =
      (which == 0 ? Wtq : (which == 1 ? Wtk : Wtv)) + (size_t)h * DH * DDIM;
  const float* bias = (which == 0 ? bq : (which == 1 ? bk : bv)) + h * DH;

  v8f acc[4] = {};
  if (which < 2) {
    for (int d0 = 0; d0 < DDIM; d0 += 32) {
      v16bf bf = load_b_frag(X + (size_t)s0 * DDIM + d0, DDIM);
#pragma unroll
      for (int t = 0; t < 4; t++) {
        v16bf af = load_a_frag(Wt + (size_t)(t * 16) * DDIM + d0, DDIM);
        acc[t] = wmma_bf16(af, bf, acc[t]);  // C^T tile: m=e, n=s
      }
    }
    // acc[t][r] = C[s = s0+n][e = 16t + 8g + r] -> packed store, row-major [s][DH]
    bf16_t* out = (which == 0 ? Qo : Ko) +
                  (((size_t)b * NH + h) * SQL + (s0 + n)) * DH;
#pragma unroll
    for (int t = 0; t < 4; t++) {
      int e0 = t * 16 + g * 8;
      v8bf pk;
#pragma unroll
      for (int r = 0; r < 8; r++) pk[r] = (bf16_t)(acc[t][r] + bias[e0 + r]);
      *(v8bf*)(out + e0) = pk;
    }
  } else {
    for (int d0 = 0; d0 < DDIM; d0 += 32) {
      v16bf af = load_a_frag(X + (size_t)s0 * DDIM + d0, DDIM);
#pragma unroll
      for (int t = 0; t < 4; t++) {
        v16bf bf = load_b_frag(Wt + (size_t)(t * 16) * DDIM + d0, DDIM);
        acc[t] = wmma_bf16(af, bf, acc[t]);  // C tile: m=s, n=e
      }
    }
    // acc[t][r] = C[s = s0+8g+r][e = 16t+n] -> packed store into V^T [dh][s]
#pragma unroll
    for (int t = 0; t < 4; t++) {
      int e = t * 16 + n;
      float bb = bias[e];
      v8bf pk;
#pragma unroll
      for (int r = 0; r < 8; r++) pk[r] = (bf16_t)(acc[t][r] + bb);
      *(v8bf*)(Vto + (((size_t)b * NH + h) * DH + e) * SKVL + s0 + g * 8) = pk;
    }
  }
}

// ---------------- flash attention core ----------------
// grid (SQL/64, NH, BB), block 128 = 4 waves; each wave owns 16 query rows.
// K/V tiles staged in LDS, shared by all 4 waves; TDM double-buffered when available.
__global__ __launch_bounds__(128) void attn(
    const bf16_t* __restrict__ Q, const bf16_t* __restrict__ K,
    const bf16_t* __restrict__ Vt, bf16_t* __restrict__ heads) {
  __shared__ bf16_t Kt[2][32 * DH];  // [kv][dh] tile(s)
  __shared__ bf16_t Vs[2][DH * 32];  // [dh][kv] tile(s)
  __shared__ bf16_t Pl[4][16 * 32];  // per-wave probability tile (A-layout staging)
  __shared__ bf16_t Ot[4][16 * 72];  // per-wave O transpose staging (padded rows)

  int wave = threadIdx.x >> 5;
  int l = threadIdx.x & 31, n = l & 15, g = l >> 4;
  int b = blockIdx.z, h = blockIdx.y;
  int s0 = blockIdx.x * 64 + wave * 16;

  const bf16_t* Qp = Q + (((size_t)b * NH + h) * SQL + s0) * DH;
  const bf16_t* Kb = K + ((size_t)b * NH + h) * SKVL * DH;
  const bf16_t* Vb = Vt + ((size_t)b * NH + h) * DH * SKVL;

  v16bf q0 = load_a_frag(Qp, DH);       // dh 0..31
  v16bf q1 = load_a_frag(Qp + 32, DH);  // dh 32..63

  v8f o[4] = {};
  float mx[8], ls[8];
#pragma unroll
  for (int r = 0; r < 8; r++) { mx[r] = -1e30f; ls[r] = 0.f; }

  bf16_t* Pw = Pl[wave];
  const float scl = 0.125f;  // 1/sqrt(64)
  const int NIT = SKVL / 32;

#if USE_TDM
  if (wave == 0) {  // prologue: fill buffer 0
    tdm_2d_load(lds_off(&Kt[0][0]), Kb, DH, SKVL, DH, DH, 32);
    tdm_2d_load(lds_off(&Vs[0][0]), Vb, SKVL, DH, SKVL, 32, DH);
  }
#endif

  for (int it = 0; it < NIT; ++it) {
    int kv0 = it * 32;
#if USE_TDM
    int buf = it & 1;
    if (wave == 0) {
      if (it + 1 < NIT) {
        // issue next tile into the other buffer, then wait for current tile
        tdm_2d_load(lds_off(&Kt[buf ^ 1][0]), Kb + (size_t)(kv0 + 32) * DH,
                    DH, SKVL, DH, DH, 32);
        tdm_2d_load(lds_off(&Vs[buf ^ 1][0]), Vb + (kv0 + 32),
                    SKVL, DH, SKVL, 32, DH);
        __builtin_amdgcn_s_wait_tensorcnt(2);
      } else {
        __builtin_amdgcn_s_wait_tensorcnt(0);
      }
    }
    __syncthreads();  // tile `buf` visible to all waves
#else
    int buf = 0;
    __syncthreads();  // previous iteration's readers done (WAR)
    {
      // cooperative fill: 128 threads x 2 x (16B K + 16B V)
      int t = threadIdx.x;
#pragma unroll
      for (int i = 0; i < 2; ++i) {
        int c = t + 128 * i;  // 256 chunks of 8 bf16
        *(v8bf*)&Kt[0][c * 8] = *(const v8bf*)(Kb + (size_t)kv0 * DH + c * 8);
        int row = c >> 2, cc = c & 3;
        *(v8bf*)&Vs[0][row * 32 + cc * 8] =
            *(const v8bf*)(Vb + (size_t)row * SKVL + kv0 + cc * 8);
      }
      if (kv0 + 32 < SKVL)  // prefetch next K tile
        __builtin_prefetch(Kb + (size_t)(kv0 + 32) * DH, 0, 1);
    }
    __syncthreads();
#endif

    v8f sc[2] = {};
#pragma unroll
    for (int j = 0; j < 2; j++) {
      v16bf k0 = load_b_frag(&Kt[buf][(16 * j) * DH], DH);       // dh 0..31
      v16bf k1 = load_b_frag(&Kt[buf][(16 * j) * DH + 32], DH);  // dh 32..63
      sc[j] = wmma_bf16(q0, k0, sc[j]);
      sc[j] = wmma_bf16(q1, k1, sc[j]);
    }
    // Online softmax. Row m = 8g + r lives in vgpr r across the 16 lanes of half g.
#pragma unroll
    for (int r = 0; r < 8; r++) {
      float v0 = sc[0][r] * scl;
      float v1 = sc[1][r] * scl;
      float rowm = fmaxf(v0, v1);
#pragma unroll
      for (int off = 8; off >= 1; off >>= 1)
        rowm = fmaxf(rowm, __shfl_xor(rowm, off, 32));
      float newm = fmaxf(mx[r], rowm);
      float corr = __expf(mx[r] - newm);
      float p0 = __expf(v0 - newm);
      float p1 = __expf(v1 - newm);
      float rs = p0 + p1;
#pragma unroll
      for (int off = 8; off >= 1; off >>= 1) rs += __shfl_xor(rs, off, 32);
      ls[r] = ls[r] * corr + rs;
      mx[r] = newm;
#pragma unroll
      for (int t = 0; t < 4; t++) o[t][r] *= corr;
      int m = g * 8 + r;
      Pw[m * 32 + n] = (bf16_t)p0;
      Pw[m * 32 + 16 + n] = (bf16_t)p1;
    }
    __syncthreads();
    v16bf pf = load_a_frag(Pw, 32);  // P tile as A operand (k = 32 kv)
#pragma unroll
    for (int t = 0; t < 4; t++) {
      v16bf vb = load_b_frag(&Vs[buf][(16 * t) * 32], 32);
      o[t] = wmma_bf16(pf, vb, o[t]);  // O[m=q][n = 16t + lane]
    }
    __syncthreads();  // release buf + Pw
  }

  // Normalize, transpose through LDS, store heads[s][h*64+dh] row-major bf16.
#pragma unroll
  for (int r = 0; r < 8; r++) {
    float inv = 1.0f / ls[r];
#pragma unroll
    for (int t = 0; t < 4; t++)
      Ot[wave][(g * 8 + r) * 72 + t * 16 + n] = (bf16_t)(o[t][r] * inv);
  }
  __syncthreads();
  bf16_t* hp = heads + ((size_t)b * SQL + s0) * DDIM + h * DH;
#pragma unroll
  for (int i = 0; i < 4; i++) {
    int c = l + 32 * i;  // 128 16-byte chunks: 16 rows x 8 chunks
    int row = c >> 3;
    int cc = c & 7;
    v8bf v = *(const v8bf*)&Ot[wave][row * 72 + cc * 8];
    *(v8bf*)(hp + (size_t)row * DDIM + cc * 8) = v;
  }
}

// ---------------- output projection (WMMA GEMM, f32 out) ----------------
// Transposed orientation: A = Wot rows [e][d], B = heads rows [s][d].
__global__ __launch_bounds__(128) void out_proj(
    const bf16_t* __restrict__ heads, const bf16_t* __restrict__ Wot,
    const float* __restrict__ bo, float* __restrict__ out) {
  int wave = threadIdx.x >> 5;
  int l = threadIdx.x & 31, n = l & 15, g = l >> 4;
  int e_base = blockIdx.x * 256 + wave * 64;
  int s0 = blockIdx.y * 16;

  v8f acc[4] = {};
  for (int d0 = 0; d0 < DDIM; d0 += 32) {
    v16bf bf = load_b_frag(heads + (size_t)s0 * DDIM + d0, DDIM);
#pragma unroll
    for (int t = 0; t < 4; t++) {
      v16bf af = load_a_frag(Wot + (size_t)(e_base + t * 16) * DDIM + d0, DDIM);
      acc[t] = wmma_bf16(af, bf, acc[t]);
    }
  }
  // acc[t][r] = C[s = s0+n][e = e_base + 16t + 8g + r] -> packed f32 stores
  float* op = out + (size_t)(s0 + n) * DDIM + e_base;
#pragma unroll
  for (int t = 0; t < 4; t++) {
    int e0 = t * 16 + g * 8;
    v4f x0, x1;
#pragma unroll
    for (int r = 0; r < 4; r++) x0[r] = acc[t][r] + bo[e_base + e0 + r];
#pragma unroll
    for (int r = 0; r < 4; r++) x1[r] = acc[t][r + 4] + bo[e_base + e0 + 4 + r];
    *(v4f*)(op + e0) = x0;
    *(v4f*)(op + e0 + 4) = x1;
  }
}

extern "C" void kernel_launch(void* const* d_in, const int* in_sizes, int n_in,
                              void* d_out, int out_size, void* d_ws, size_t ws_size,
                              hipStream_t stream) {
  (void)in_sizes; (void)n_in; (void)out_size; (void)ws_size;
  const float* query = (const float*)d_in[0];
  const float* value = (const float*)d_in[1];
  const float* wq = (const float*)d_in[2];
  const float* bq = (const float*)d_in[3];
  const float* wk = (const float*)d_in[4];
  const float* bk = (const float*)d_in[5];
  const float* wv = (const float*)d_in[6];
  const float* bv = (const float*)d_in[7];
  const float* wo = (const float*)d_in[8];
  const float* bo = (const float*)d_in[9];
  float* out = (float*)d_out;

  // Workspace carve-out (~104 MB total)
  char* ws = (char*)d_ws;
  size_t off = 0;
  auto alloc = [&](size_t bytes) -> char* {
    char* p = ws + off;
    off += (bytes + 255) & ~(size_t)255;
    return p;
  };
  bf16_t* qbf = (bf16_t*)alloc((size_t)BB * SQL * DDIM * 2);
  bf16_t* vbf = (bf16_t*)alloc((size_t)BB * SKVL * DDIM * 2);
  bf16_t* Wtq = (bf16_t*)alloc((size_t)NH * DH * DDIM * 2);
  bf16_t* Wtk = (bf16_t*)alloc((size_t)NH * DH * DDIM * 2);
  bf16_t* Wtv = (bf16_t*)alloc((size_t)NH * DH * DDIM * 2);
  bf16_t* Wot = (bf16_t*)alloc((size_t)DDIM * DDIM * 2);
  bf16_t* Qb  = (bf16_t*)alloc((size_t)BB * NH * SQL * DH * 2);
  bf16_t* Kb  = (bf16_t*)alloc((size_t)BB * NH * SKVL * DH * 2);
  bf16_t* Vtb = (bf16_t*)alloc((size_t)BB * NH * DH * SKVL * 2);
  bf16_t* Hd  = (bf16_t*)alloc((size_t)BB * SQL * DDIM * 2);

  int nq = BB * SQL * DDIM;
  int nv = BB * SKVL * DDIM;
  cvt_f32_bf16<<<(nq + 255) / 256, 256, 0, stream>>>(query, qbf, nq);
  cvt_f32_bf16<<<(nv + 255) / 256, 256, 0, stream>>>(value, vbf, nv);

  int nw = NH * DDIM * DH;
  transpose_w<<<(nw + 255) / 256, 256, 0, stream>>>(wq, Wtq, NH, DDIM, DH);
  transpose_w<<<(nw + 255) / 256, 256, 0, stream>>>(wk, Wtk, NH, DDIM, DH);
  transpose_w<<<(nw + 255) / 256, 256, 0, stream>>>(wv, Wtv, NH, DDIM, DH);
  int nwo = DDIM * DDIM;
  transpose_w<<<(nwo + 255) / 256, 256, 0, stream>>>(wo, Wot, 1, DDIM, DDIM);

  qkv_proj<<<dim3(SQL / 64, NH, BB * 3), 128, 0, stream>>>(
      qbf, vbf, Wtq, Wtk, Wtv, bq, bk, bv, Qb, Kb, Vtb);

  attn<<<dim3(SQL / 64, NH, BB), 128, 0, stream>>>(Qb, Kb, Vtb, Hd);

  out_proj<<<dim3(DDIM / 256, BB * SQL / 16), 128, 0, stream>>>(Hd, Wot, bo, out);
}
